// SAMamba_76020921140264
// MI455X (gfx1250) — matrix operand order repaired and checked by reading, not compiled, not run on hardware
//
#include <hip/hip_runtime.h>
#include <stdint.h>

typedef __attribute__((ext_vector_type(8)))  float        v8f;
typedef __attribute__((ext_vector_type(16))) __bf16       v16bf;
typedef __attribute__((ext_vector_type(8)))  unsigned int v8u;

#define DEV __device__ __forceinline__

#if __has_builtin(__builtin_amdgcn_global_load_async_to_lds_b128) && \
    __has_builtin(__builtin_amdgcn_s_wait_asynccnt)
#define USE_ASYNC_LDS 1
#else
#define USE_ASYNC_LDS 0
#endif

DEV unsigned short f2bfbits(float f) {
  unsigned int x = __float_as_uint(f);
  unsigned int r = x + 0x7FFFu + ((x >> 16) & 1u);   // round-to-nearest-even
  return (unsigned short)(r >> 16);
}
DEV float bf2f(unsigned short u) { return __uint_as_float(((unsigned int)u) << 16); }
DEV float sigmoidf_(float x)     { return 1.f / (1.f + __expf(-x)); }

// two f32 -> packed bf16x2 in one u32 (hardware cvt if present, else perm pack)
DEV unsigned int pack2f(float x, float y) {
#if __has_builtin(__builtin_amdgcn_cvt_pk_bf16_f32)
  return __builtin_bit_cast(unsigned int, __builtin_amdgcn_cvt_pk_bf16_f32(x, y));
#else
  unsigned int a = __float_as_uint(x), b = __float_as_uint(y);
  a = a + 0x7FFFu + ((a >> 16) & 1u);
  b = b + 0x7FFFu + ((b >> 16) & 1u);
  return __builtin_amdgcn_perm(b, a, 0x07060302u);   // {b.hi16, a.hi16}
#endif
}

// copy one 32-element bf16 row chunk (64B) into two LDS half-tiles
DEV void fill_row(const unsigned short* src, unsigned short* d0, unsigned short* d1) {
#if USE_ASYNC_LDS
  typedef int vi4 __attribute__((__vector_size__(16)));
  typedef __attribute__((address_space(1))) vi4* gp_t;   // global int4*
  typedef __attribute__((address_space(3))) vi4* lp_t;   // LDS int4*
  __builtin_amdgcn_global_load_async_to_lds_b128((gp_t)(src),      (lp_t)(d0),     0, 0);
  __builtin_amdgcn_global_load_async_to_lds_b128((gp_t)(src + 8),  (lp_t)(d0 + 8), 0, 0);
  __builtin_amdgcn_global_load_async_to_lds_b128((gp_t)(src + 16), (lp_t)(d1),     0, 0);
  __builtin_amdgcn_global_load_async_to_lds_b128((gp_t)(src + 24), (lp_t)(d1 + 8), 0, 0);
  __builtin_amdgcn_s_wait_asynccnt(0);
#else
  const uint4* s4 = (const uint4*)src;
  ((uint4*)d0)[0] = s4[0]; ((uint4*)d0)[1] = s4[1];
  ((uint4*)d1)[0] = s4[2]; ((uint4*)d1)[1] = s4[3];
#endif
}

// one-shot f32 -> bf16 weight conversion
__global__ void k_cvt_bf16(const float* __restrict__ in,
                           unsigned short* __restrict__ out, int n)
{
  int i = blockIdx.x * 256 + threadIdx.x;
  if (i < n) out[i] = f2bfbits(in[i]);
}

// ---------------------------------------------------------------------------
// K1: Y = bf16(relu(X(f32, MxK) @ Wbf(256xK)^T + b)); wave = 16x256 stripe.
// B tile (256x32 bf16 = 16KB) staged in LDS per block, shared by 8 waves.
// LDS layout: sB[h*4096 + row*16 + j] holds W[row, kt + h*16 + j]
// ---------------------------------------------------------------------------
__global__ __launch_bounds__(256) void k_fc_f32_wmma(
    const float* __restrict__ X, const unsigned short* __restrict__ Wbf,
    const float* __restrict__ bias, unsigned short* __restrict__ Out,
    int M, int K)
{
  __shared__ unsigned short sB[8192];
  const int tid  = threadIdx.x;
  const int lane = tid & 31;
  const int wave = blockIdx.x * 8 + (tid >> 5);
  const int m0   = wave << 4;
  const int rsub = lane & 15;
  const int hi   = lane >> 4;
  const bool active = m0 < M;
  int row = m0 + rsub;
  int rclamp = row < M ? row : M - 1;
  const float* xr = X + (size_t)rclamp * K;

  v8f acc[16];
#pragma unroll
  for (int nt = 0; nt < 16; ++nt)
#pragma unroll
    for (int j = 0; j < 8; ++j) acc[nt][j] = 0.f;

  for (int kt = 0; kt < K; kt += 32) {
    __syncthreads();
    fill_row(Wbf + (size_t)tid * K + kt, sB + tid * 16, sB + 4096 + tid * 16);
    __syncthreads();
    if (active) {
      int ka0 = kt + (hi ? 16 : 0);
      int ka1 = kt + (hi ? 24 : 8);
      v8u aw;
#pragma unroll
      for (int j = 0; j < 4; ++j) aw[j]     = pack2f(xr[ka0 + 2 * j], xr[ka0 + 2 * j + 1]);
#pragma unroll
      for (int j = 0; j < 4; ++j) aw[4 + j] = pack2f(xr[ka1 + 2 * j], xr[ka1 + 2 * j + 1]);
      v16bf a = __builtin_bit_cast(v16bf, aw);
      if (kt + 32 < K) __builtin_prefetch(xr + kt + 64, 0, 0);
#pragma unroll
      for (int nt = 0; nt < 16; ++nt) {
        v16bf b = *(const v16bf*)(sB + hi * 4096 + (nt * 16 + rsub) * 16);
        acc[nt] = __builtin_amdgcn_wmma_f32_16x16x32_bf16(
            false, a, false, b, (short)0, acc[nt], false, false);
      }
    }
  }
  if (active) {
    const int roff = hi ? 8 : 0;
#pragma unroll
    for (int nt = 0; nt < 16; ++nt) {
      int n = nt * 16 + rsub;
      float bv = bias[n];
#pragma unroll
      for (int r = 0; r < 8; ++r) {
        int m = m0 + roff + r;
        if (m < M) {
          float v = acc[nt][r] + bv;
          Out[(size_t)m * 256 + n] = f2bfbits(v > 0.f ? v : 0.f);
        }
      }
    }
  }
}

// ---------------------------------------------------------------------------
// K2: same but bf16 input (in-place safe: wave reads only its own rows)
// ---------------------------------------------------------------------------
__global__ __launch_bounds__(256) void k_fc_bf16_wmma(
    const unsigned short* Xbf, const unsigned short* __restrict__ Wbf,
    const float* __restrict__ bias, unsigned short* Out, int M, int K)
{
  __shared__ unsigned short sB[8192];
  const int tid  = threadIdx.x;
  const int lane = tid & 31;
  const int wave = blockIdx.x * 8 + (tid >> 5);
  const int m0   = wave << 4;
  const int rsub = lane & 15;
  const int hi   = lane >> 4;
  const bool active = m0 < M;
  int row = m0 + rsub;
  int rclamp = row < M ? row : M - 1;
  const unsigned short* xr = Xbf + (size_t)rclamp * K;

  v8f acc[16];
#pragma unroll
  for (int nt = 0; nt < 16; ++nt)
#pragma unroll
    for (int j = 0; j < 8; ++j) acc[nt][j] = 0.f;

  for (int kt = 0; kt < K; kt += 32) {
    __syncthreads();
    fill_row(Wbf + (size_t)tid * K + kt, sB + tid * 16, sB + 4096 + tid * 16);
    __syncthreads();
    if (active) {
      int ka0 = kt + (hi ? 16 : 0);
      int ka1 = kt + (hi ? 24 : 8);
      uint4 u0 = *(const uint4*)(xr + ka0);
      uint4 u1 = *(const uint4*)(xr + ka1);
      v8u aw;
      aw[0] = u0.x; aw[1] = u0.y; aw[2] = u0.z; aw[3] = u0.w;
      aw[4] = u1.x; aw[5] = u1.y; aw[6] = u1.z; aw[7] = u1.w;
      v16bf a = __builtin_bit_cast(v16bf, aw);
#pragma unroll
      for (int nt = 0; nt < 16; ++nt) {
        v16bf b = *(const v16bf*)(sB + hi * 4096 + (nt * 16 + rsub) * 16);
        acc[nt] = __builtin_amdgcn_wmma_f32_16x16x32_bf16(
            false, a, false, b, (short)0, acc[nt], false, false);
      }
    }
  }
  if (active) {
    const int roff = hi ? 8 : 0;
#pragma unroll
    for (int nt = 0; nt < 16; ++nt) {
      int n = nt * 16 + rsub;
      float bv = bias[n];
#pragma unroll
      for (int r = 0; r < 8; ++r) {
        int m = m0 + roff + r;
        if (m < M) {
          float v = acc[nt][r] + bv;
          Out[(size_t)m * 256 + n] = f2bfbits(v > 0.f ? v : 0.f);
        }
      }
    }
  }
}

// ---------------------------------------------------------------------------
// K3: per-row sum of squares of bf16 h
// ---------------------------------------------------------------------------
__global__ void k_rowsumsq(const unsigned short* __restrict__ Hbf,
                           float* __restrict__ rn2, int M)
{
  int m = blockIdx.x * 256 + threadIdx.x;
  if (m >= M) return;
  const uint4* r = (const uint4*)(Hbf + (size_t)m * 256);
  float s = 0.f;
#pragma unroll 4
  for (int i = 0; i < 32; ++i) {
    uint4 u = r[i];
    unsigned int w[4] = {u.x, u.y, u.z, u.w};
#pragma unroll
    for (int q = 0; q < 4; ++q) {
      float a = bf2f((unsigned short)(w[q] & 0xffffu));
      float b = bf2f((unsigned short)(w[q] >> 16));
      s += a * a + b * b;
    }
  }
  rn2[m] = s;
}

__global__ void k_initkeys(unsigned long long* keys) { keys[threadIdx.x] = ~0ull; }

// ---------------------------------------------------------------------------
// K4: S[m,n] = ||h_m||^2 - 2 h_m . c_n via WMMA; argmin over m per column n.
// key = (monotonic_float(S) << 32) | m -> LDS then global atomicMin.
// ---------------------------------------------------------------------------
__global__ __launch_bounds__(256) void k_cluster_wmma(
    const unsigned short* __restrict__ Hbf, const unsigned short* __restrict__ Cbf,
    const float* __restrict__ rn2, unsigned long long* __restrict__ gkeys, int M)
{
  __shared__ unsigned long long sk[256];
  __shared__ unsigned short sB[8192];
  const int tid = threadIdx.x;
  sk[tid] = ~0ull;

  const int lane = tid & 31;
  const int wave = blockIdx.x * 8 + (tid >> 5);
  const int m0   = wave << 4;
  const int rsub = lane & 15;
  const int hi   = lane >> 4;
  const bool active = m0 < M;
  int row = m0 + rsub;
  int rclamp = row < M ? row : M - 1;
  const unsigned short* xr = Hbf + (size_t)rclamp * 256;

  v8f acc[16];
#pragma unroll
  for (int nt = 0; nt < 16; ++nt)
#pragma unroll
    for (int j = 0; j < 8; ++j) acc[nt][j] = 0.f;

  for (int kt = 0; kt < 256; kt += 32) {
    __syncthreads();
    fill_row(Cbf + (size_t)tid * 256 + kt, sB + tid * 16, sB + 4096 + tid * 16);
    __syncthreads();
    if (active) {
      int ka0 = kt + (hi ? 16 : 0);
      int ka1 = kt + (hi ? 24 : 8);
      uint4 u0 = *(const uint4*)(xr + ka0);
      uint4 u1 = *(const uint4*)(xr + ka1);
      v8u aw;
      aw[0] = u0.x; aw[1] = u0.y; aw[2] = u0.z; aw[3] = u0.w;
      aw[4] = u1.x; aw[5] = u1.y; aw[6] = u1.z; aw[7] = u1.w;
      v16bf a = __builtin_bit_cast(v16bf, aw);
#pragma unroll
      for (int nt = 0; nt < 16; ++nt) {
        v16bf b = *(const v16bf*)(sB + hi * 4096 + (nt * 16 + rsub) * 16);
        acc[nt] = __builtin_amdgcn_wmma_f32_16x16x32_bf16(
            false, a, false, b, (short)0, acc[nt], false, false);
      }
    }
  }

  if (active) {
    const int roff = hi ? 8 : 0;
#pragma unroll
    for (int nt = 0; nt < 16; ++nt) {
      unsigned long long best = ~0ull;
#pragma unroll
      for (int r = 0; r < 8; ++r) {
        int m = m0 + roff + r;
        if (m < M) {
          float s = rn2[m] - 2.f * acc[nt][r];
          unsigned int u = __float_as_uint(s);
          u = (u & 0x80000000u) ? ~u : (u | 0x80000000u);
          unsigned long long key = ((unsigned long long)u << 32) | (unsigned int)m;
          best = key < best ? key : best;
        }
      }
      unsigned int lo = (unsigned int)best, hw = (unsigned int)(best >> 32);
      unsigned int lo2 = __shfl_xor(lo, 16), hw2 = __shfl_xor(hw, 16);
      unsigned long long other = ((unsigned long long)hw2 << 32) | lo2;
      best = other < best ? other : best;
      if (lane < 16) atomicMin(&sk[nt * 16 + rsub], best);
    }
  }
  __syncthreads();
  atomicMin(&gkeys[tid], sk[tid]);
}

__global__ void k_gather(const unsigned long long* __restrict__ keys,
                         const unsigned short* __restrict__ Hbf,
                         float* __restrict__ out)
{
  int n = blockIdx.x, t = threadIdx.x;
  int m = (int)(keys[n] & 0xffffffffull);
  out[(size_t)n * 256 + t] = bf2f(Hbf[(size_t)m * 256 + t]);
}

// ---------------------------------------------------------------------------
// Generic small scalar GEMM: out = act(scale*(A@B') + bias)
// transB=1: B is (N,K) row-major (x @ W.T); transB=0: B is (K,N)
// act: 0 none, 1 relu, 2 silu, 3 softplus, 4 sigmoid, 5 elu
// ---------------------------------------------------------------------------
__global__ void k_sgemm(const float* __restrict__ A, const float* __restrict__ B,
                        const float* __restrict__ bias, float* __restrict__ out,
                        int rows, int N, int K, int lda, int transB,
                        float scale, int act)
{
  int idx = blockIdx.x * 256 + threadIdx.x;
  if (idx >= rows * N) return;
  int r = idx / N, n = idx % N;
  const float* a = A + (size_t)r * lda;
  float s = 0.f;
  if (transB) {
    const float* b = B + (size_t)n * K;
    for (int k = 0; k < K; ++k) s += a[k] * b[k];
  } else {
    for (int k = 0; k < K; ++k) s += a[k] * B[(size_t)k * N + n];
  }
  s *= scale;
  if (bias) s += bias[n];
  switch (act) {
    case 1: s = s > 0.f ? s : 0.f; break;
    case 2: s = s * sigmoidf_(s); break;
    case 3: s = (s > 20.f) ? s : log1pf(__expf(s)); break;
    case 4: s = sigmoidf_(s); break;
    case 5: s = s > 0.f ? s : (__expf(s) - 1.f); break;
    default: break;
  }
  out[(size_t)r * N + n] = s;
}

// causal depthwise conv (k=4, left pad 3) + bias + silu; xh = xz[:, :512]
__global__ void k_conv_silu(const float* __restrict__ xz, const float* __restrict__ cW,
                            const float* __restrict__ cb, float* __restrict__ xc, int L)
{
  int idx = blockIdx.x * 256 + threadIdx.x;
  if (idx >= L * 512) return;
  int t = idx / 512, d = idx % 512;
  float s = cb[d];
#pragma unroll
  for (int j = 0; j < 4; ++j) {
    int tt = t - 3 + j;
    if (tt >= 0) s += cW[d * 4 + j] * xz[(size_t)tt * 1024 + d];
  }
  xc[idx] = s * sigmoidf_(s);
}

// selective scan: one block of 512 threads (one per channel), 16 states in regs
__global__ __launch_bounds__(512) void k_scan(
    const float* __restrict__ dt, const float* __restrict__ dbl,
    const float* __restrict__ xc, const float* __restrict__ A_log,
    float* __restrict__ ys, int L)
{
  int d = threadIdx.x;
  float A[16], h[16];
#pragma unroll
  for (int s = 0; s < 16; ++s) { A[s] = -__expf(A_log[d * 16 + s]); h[s] = 0.f; }
  __shared__ float Bs[16], Cs[16];
  for (int t = 0; t < L; ++t) {
    __syncthreads();
    if (d < 16)       Bs[d]      = dbl[t * 48 + 16 + d];
    else if (d < 32)  Cs[d - 16] = dbl[t * 48 + 32 + (d - 16)];
    __syncthreads();
    float dtv = dt[(size_t)t * 512 + d], xv = xc[(size_t)t * 512 + d];
    float y = 0.f;
#pragma unroll
    for (int s = 0; s < 16; ++s) {
      h[s] = __expf(dtv * A[s]) * h[s] + dtv * Bs[s] * xv;
      y += h[s] * Cs[s];
    }
    ys[(size_t)t * 512 + d] = y;
  }
}

// y = (ys + xc*D) * silu(z), z = xz[:, 512:]
__global__ void k_gate(const float* __restrict__ ys, const float* __restrict__ xc,
                       const float* __restrict__ xz, const float* __restrict__ Dv,
                       float* __restrict__ out, int L)
{
  int idx = blockIdx.x * 256 + threadIdx.x;
  if (idx >= L * 512) return;
  int t = idx / 512, d = idx % 512;
  float z = xz[(size_t)t * 1024 + 512 + d];
  float y = ys[idx] + xc[idx] * Dv[d];
  out[idx] = y * (z * sigmoidf_(z));
}

// row softmax, len <= 256, one block per row
__global__ __launch_bounds__(256) void k_softmax(float* buf, int len)
{
  float* p = buf + (size_t)blockIdx.x * len;
  int t = threadIdx.x;
  __shared__ float red[256];
  float v = (t < len) ? p[t] : -3.4e38f;
  red[t] = v; __syncthreads();
  for (int s = 128; s > 0; s >>= 1) { if (t < s) red[t] = fmaxf(red[t], red[t + s]); __syncthreads(); }
  float mx = red[0]; __syncthreads();
  float e = (t < len) ? __expf(v - mx) : 0.f;
  red[t] = e; __syncthreads();
  for (int s = 128; s > 0; s >>= 1) { if (t < s) red[t] += red[t + s]; __syncthreads(); }
  float sum = red[0];
  if (t < len) p[t] = e / sum;
}

__global__ void k_final(const float* __restrict__ logits, float* __restrict__ out)
{
  if (threadIdx.x == 0) {
    float S = 1.f;
    for (int i = 0; i < 4; ++i) out[i] = sigmoidf_(logits[i]);
    for (int i = 0; i < 4; ++i) { S *= (1.f - out[i]); out[4 + i] = S; }
  }
}

// ---------------------------------------------------------------------------
extern "C" void kernel_launch(void* const* d_in, const int* in_sizes, int n_in,
                              void* d_out, int out_size, void* d_ws, size_t ws_size,
                              hipStream_t stream)
{
  (void)n_in; (void)out_size; (void)ws_size;
  auto F = [&](int i) { return (const float*)d_in[i]; };
  // flattened input order: x_path, x_omics[0..5], then params dict in insertion order
  const float* X = F(0);
  const float* xom[6]; for (int i = 0; i < 6; ++i) xom[i] = F(1 + i);
  const float *pfc_W0 = F(7), *pfc_b0 = F(8), *pfc_W1 = F(9), *pfc_b1 = F(10);
  const float* centers = F(11);
  const float *gW0[6], *gb0[6], *gW1[6], *gb1[6];
  for (int i = 0; i < 6; ++i) {
    gW0[i] = F(12 + 4 * i); gb0[i] = F(13 + 4 * i);
    gW1[i] = F(14 + 4 * i); gb1[i] = F(15 + 4 * i);
  }
  struct MP { const float *in_proj, *conv_W, *conv_b, *x_proj, *dt_W, *dt_b, *A_log, *D, *out_proj; };
  auto mp = [&](int b) {
    MP m; m.in_proj = F(b); m.conv_W = F(b + 1); m.conv_b = F(b + 2); m.x_proj = F(b + 3);
    m.dt_W = F(b + 4); m.dt_b = F(b + 5); m.A_log = F(b + 6); m.D = F(b + 7); m.out_proj = F(b + 8);
    return m;
  };
  MP g_enc = mp(36), g_dec = mp(45), p_enc = mp(54), p_dec = mp(63);
  struct AP { const float *in_W, *in_b, *out_W, *out_b; };
  AP pg{F(72), F(73), F(74), F(75)}, gp{F(76), F(77), F(78), F(79)};
  struct PP { const float *q, *W, *b; };
  PP ppool{F(80), F(81), F(82)}, gpool{F(83), F(84), F(85)};
  const float *mmW0 = F(86), *mmb0 = F(87), *mmW1 = F(88), *mmb1 = F(89);
  const float *clsW = F(90), *clsb = F(91);

  const int M = 100000;
  char* base = (char*)d_ws; size_t off = 0;
  auto alloc = [&](size_t bytes) -> void* {
    off = (off + 255) & ~(size_t)255; void* p = base + off; off += bytes; return p;
  };
  unsigned short* Hbf = (unsigned short*)alloc((size_t)M * 256 * 2); // fc1 out, then fc2 in-place
  unsigned short* wbf1 = (unsigned short*)alloc(256 * 1024 * 2);
  unsigned short* wbf2 = (unsigned short*)alloc(256 * 256 * 2);
  unsigned short* cbf  = (unsigned short*)alloc(256 * 256 * 2);
  float* rn2 = (float*)alloc((size_t)M * 4);
  unsigned long long* keys = (unsigned long long*)alloc(256 * 8);
  float* path_feats = (float*)alloc(256 * 256 * 4);
  float* gene1 = (float*)alloc(6 * 1024 * 4);
  float* gene_feats = (float*)alloc(6 * 256 * 4);
  float* xz  = (float*)alloc(256 * 1024 * 4);
  float* xcb = (float*)alloc(256 * 512 * 4);
  float* dbl = (float*)alloc(256 * 48 * 4);
  float* dtb = (float*)alloc(256 * 512 * 4);
  float* ysb = (float*)alloc(256 * 512 * 4);
  float* yact = (float*)alloc(256 * 512 * 4);
  float* path_enc = (float*)alloc(256 * 256 * 4);
  float* gene_enc = (float*)alloc(6 * 256 * 4);
  float* p_in_g = (float*)alloc(256 * 256 * 4);
  float* g_in_p = (float*)alloc(6 * 256 * 4);
  float* pdec = (float*)alloc(256 * 256 * 4);
  float* gdec = (float*)alloc(6 * 256 * 4);
  float* qb = (float*)alloc(256 * 256 * 4);
  float* kb = (float*)alloc(256 * 256 * 4);
  float* vb = (float*)alloc(256 * 256 * 4);
  float* sc = (float*)alloc(256 * 256 * 4);
  float* ob = (float*)alloc(256 * 256 * 4);
  float* xp = (float*)alloc(256 * 256 * 4);
  float* sv = (float*)alloc(256 * 4);
  float* fbuf = (float*)alloc(512 * 4);
  float* mm1 = (float*)alloc(256 * 4);
  float* mm2 = (float*)alloc(256 * 4);
  float* logits = (float*)alloc(4 * 4);

  auto sgemm = [&](const float* A, const float* B, const float* bias, float* out,
                   int rows, int N, int K, int lda, int transB, float scale, int act) {
    int tot = rows * N;
    k_sgemm<<<dim3((tot + 255) / 256), dim3(256), 0, stream>>>(
        A, B, bias, out, rows, N, K, lda, transB, scale, act);
  };

  // ---- weight pre-conversion to bf16 ----
  k_cvt_bf16<<<dim3((256 * 1024 + 255) / 256), dim3(256), 0, stream>>>(pfc_W0, wbf1, 256 * 1024);
  k_cvt_bf16<<<dim3((256 * 256 + 255) / 256), dim3(256), 0, stream>>>(pfc_W1, wbf2, 256 * 256);
  k_cvt_bf16<<<dim3((256 * 256 + 255) / 256), dim3(256), 0, stream>>>(centers, cbf, 256 * 256);

  // ---- path tower (WMMA heavy) ----
  int waves = (M + 15) / 16;
  int blocks = (waves + 7) / 8;
  k_fc_f32_wmma<<<dim3(blocks), dim3(256), 0, stream>>>(X, wbf1, pfc_b0, Hbf, M, 1024);
  k_fc_bf16_wmma<<<dim3(blocks), dim3(256), 0, stream>>>(Hbf, wbf2, pfc_b1, Hbf, M, 256);
  k_rowsumsq<<<dim3((M + 255) / 256), dim3(256), 0, stream>>>(Hbf, rn2, M);
  k_initkeys<<<dim3(1), dim3(256), 0, stream>>>(keys);
  k_cluster_wmma<<<dim3(blocks), dim3(256), 0, stream>>>(Hbf, cbf, rn2, keys, M);
  k_gather<<<dim3(256), dim3(256), 0, stream>>>(keys, Hbf, path_feats);

  // ---- gene tower ----
  for (int i = 0; i < 6; ++i) {
    int di = in_sizes[1 + i];
    sgemm(xom[i], gW0[i], gb0[i], gene1 + i * 1024, 1, 1024, di, di, 1, 1.f, 5);
  }
  for (int i = 0; i < 6; ++i)
    sgemm(gene1 + i * 1024, gW1[i], gb1[i], gene_feats + i * 256, 1, 256, 1024, 1024, 1, 1.f, 5);

  // ---- mamba ----
  auto mamba = [&](const float* xin, int L, const MP& P, float* out) {
    sgemm(xin, P.in_proj, nullptr, xz, L, 1024, 256, 256, 1, 1.f, 0);
    k_conv_silu<<<dim3((L * 512 + 255) / 256), dim3(256), 0, stream>>>(xz, P.conv_W, P.conv_b, xcb, L);
    sgemm(xcb, P.x_proj, nullptr, dbl, L, 48, 512, 512, 1, 1.f, 0);
    sgemm(dbl, P.dt_W, P.dt_b, dtb, L, 512, 16, 48, 1, 1.f, 3);
    k_scan<<<dim3(1), dim3(512), 0, stream>>>(dtb, dbl, xcb, P.A_log, ysb, L);
    k_gate<<<dim3((L * 512 + 255) / 256), dim3(256), 0, stream>>>(ysb, xcb, xz, P.D, yact, L);
    sgemm(yact, P.out_proj, nullptr, out, L, 256, 512, 512, 1, 1.f, 0);
  };
  mamba(gene_feats, 6, g_enc, gene_enc);
  mamba(path_feats, 256, p_enc, path_enc);

  // ---- cross attention ----
  auto mha = [&](const float* qin, int Lq, const float* kvin, int Lk, const AP& P, float* out) {
    sgemm(qin,  P.in_W,                 P.in_b,       qb, Lq, 256, 256, 256, 1, 1.f, 0);
    sgemm(kvin, P.in_W + 256 * 256,     P.in_b + 256, kb, Lk, 256, 256, 256, 1, 1.f, 0);
    sgemm(kvin, P.in_W + 2 * 256 * 256, P.in_b + 512, vb, Lk, 256, 256, 256, 1, 1.f, 0);
    sgemm(qb, kb, nullptr, sc, Lq, Lk, 256, 256, 1, 0.0625f, 0);   // 1/sqrt(256)
    k_softmax<<<dim3(Lq), dim3(256), 0, stream>>>(sc, Lk);
    sgemm(sc, vb, nullptr, ob, Lq, 256, Lk, Lk, 0, 1.f, 0);
    sgemm(ob, P.out_W, P.out_b, out, Lq, 256, 256, 256, 1, 1.f, 0);
  };
  mha(path_enc, 256, gene_enc, 6, pg, p_in_g);
  mha(gene_enc, 6, path_enc, 256, gp, g_in_p);
  mamba(p_in_g, 256, p_dec, pdec);
  mamba(g_in_p, 6, g_dec, gdec);

  // ---- pools + heads ----
  auto pool = [&](const float* x, int L, const PP& P, float* out) {
    sgemm(x, P.W, P.b, xp, L, 256, 256, 256, 1, 1.f, 0);
    sgemm(xp, P.q, nullptr, sv, L, 1, 256, 256, 1, 1.f, 0);
    k_softmax<<<dim3(1), dim3(256), 0, stream>>>(sv, L);
    sgemm(sv, x, nullptr, out, 1, 256, L, L, 0, 1.f, 0);
  };
  pool(pdec, 256, ppool, fbuf);
  pool(gdec, 6, gpool, fbuf + 256);
  sgemm(fbuf, mmW0, mmb0, mm1, 1, 256, 512, 512, 1, 1.f, 1);
  sgemm(mm1, mmW1, mmb1, mm2, 1, 256, 256, 256, 1, 1.f, 1);
  sgemm(mm2, clsW, clsb, logits, 1, 4, 256, 256, 1, 1.f, 0);
  k_final<<<dim3(1), dim3(32), 0, stream>>>(logits, (float*)d_out);
}
